// AttentionGated_81630148428072
// MI455X (gfx1250) — matrix operand order repaired and checked by reading, not compiled, not run on hardware
//
#include <hip/hip_runtime.h>

// ---------------------------------------------------------------------------
// Gated single-head attention for MI455X (gfx1250, wave32, WMMA bf16 path).
// B=4, L=2048, E=512, D=64  ->  N = B*L = 8192 tokens.
// All intermediates bf16; f32 accumulation. V kept transposed in global so
// every WMMA fragment is a contiguous 16B/32B load (no scalar gathers).
// Wave reductions use ds_swizzle_b32 (imm XOR patterns, no address setup).
// ---------------------------------------------------------------------------

#define NTOK   8192
#define EDIM   512
#define DDIM   64
#define LSEQ   2048
#define BBATCH 4
#define VPITCH 72   // bf16 elements per LDS V row (144B, 16B-aligned rows)

typedef __attribute__((ext_vector_type(16))) __bf16 v16bf;
typedef __attribute__((ext_vector_type(8)))  __bf16 v8bf;
typedef __attribute__((ext_vector_type(4)))  __bf16 v4bf;
typedef __attribute__((ext_vector_type(8)))  float  v8f;
typedef __attribute__((ext_vector_type(4)))  float  v4f;

__device__ inline __bf16 f2bf(float f) { return (__bf16)f; }   // native cvt (RNE)
__device__ inline float  bf2f(__bf16 b) { return (float)b; }

// ds_swizzle_b32 group-of-32 XOR: offset = and_mask(0x1f) | xor_mask<<10.
template <int M>
__device__ inline float swz(float x) {
  return __builtin_bit_cast(
      float, __builtin_amdgcn_ds_swizzle(__builtin_bit_cast(int, x),
                                         0x1f | (M << 10)));
}
__device__ inline float wred_sum(float x) {
  x += swz<16>(x); x += swz<8>(x); x += swz<4>(x); x += swz<2>(x); x += swz<1>(x);
  return x;
}
__device__ inline float wred_max(float x) {
  x = fmaxf(x, swz<16>(x)); x = fmaxf(x, swz<8>(x)); x = fmaxf(x, swz<4>(x));
  x = fmaxf(x, swz<2>(x));  x = fmaxf(x, swz<1>(x));
  return x;
}

__device__ inline v8f wmma_bf16(v16bf a, v16bf b, v8f c) {
  // 8 args: (neg_a, A, neg_b, B, c_mod, C, reuse_a, reuse_b)
  return __builtin_amdgcn_wmma_f32_16x16x32_bf16(false, a, false, b,
                                                 (short)0, c, false, false);
}

// A fragment (16x32 bf16), contiguous K order matching B:
// lane<16 (hf=0): elems 0..15 -> K=0..15
// lane>=16(hf=1): elems 0..15 -> K=16..31
// Two contiguous 16B chunks -> b128 loads, zero conversion VALU.
__device__ inline v16bf frag_a_bf16(const __bf16* __restrict__ rowptr, int kb, int hf) {
  v8bf lo = *(const v8bf*)(rowptr + kb + hf * 16);
  v8bf hi = *(const v8bf*)(rowptr + kb + hf * 16 + 8);
  v16bf a;
#pragma unroll
  for (int j = 0; j < 8; ++j) { a[j] = lo[j]; a[8 + j] = hi[j]; }
  return a;
}

// B fragment (32x16 bf16): lane = column N, elements = 16 contiguous K values
// (K = hf*16 + e). One 32B run -> two b128 loads.
__device__ inline v16bf frag_b_bf16(const __bf16* __restrict__ p) {
  v8bf lo = *(const v8bf*)(p);
  v8bf hi = *(const v8bf*)(p + 8);
  v16bf b;
#pragma unroll
  for (int j = 0; j < 8; ++j) { b[j] = lo[j]; b[8 + j] = hi[j]; }
  return b;
}

// ---------------------------------------------------------------------------
// fp32 -> bf16 bulk convert, 4 elements per thread.
// ---------------------------------------------------------------------------
__global__ void __launch_bounds__(256)
hconv(const float* __restrict__ x, __bf16* __restrict__ y, int n4) {
  int i = blockIdx.x * blockDim.x + threadIdx.x;
  if (i < n4) {
    v4f xv = *(const v4f*)(x + (size_t)i * 4);
    v4bf yv;
#pragma unroll
    for (int j = 0; j < 4; ++j) yv[j] = f2bf(xv[j]);
    *(v4bf*)(y + (size_t)i * 4) = yv;
  }
}

// ---------------------------------------------------------------------------
// Weight transpose + bf16 convert:  Wt[d][k] = bf16(W[k][d])
// ---------------------------------------------------------------------------
__global__ void wtrans(const float* __restrict__ W, __bf16* __restrict__ Wt,
                       int K, int Dout) {
  int d = blockIdx.x;
  for (int kk = threadIdx.x; kk < K; kk += blockDim.x)
    Wt[(size_t)d * K + kk] = f2bf(W[(size_t)kk * Dout + d]);
}

// ---------------------------------------------------------------------------
// Generic WMMA GEMM with activation epilogue.
//   C[N, COLS] = act( A[N, K] @ Wt[COLS, K]^T + bias )
// Block tile: 32 rows x COLS cols, one 16x16 tile per wave (2 x COLS/16 waves).
// ACT: 0 = none, 1 = silu, 2 = sigmoid.  TOut: __bf16 or float.
// ---------------------------------------------------------------------------
template <int ACT, int COLS, typename TOut>
__global__ void __launch_bounds__(4 * COLS)
proj_gemm(const __bf16* __restrict__ A, const __bf16* __restrict__ Wt,
          const float* __restrict__ bias, TOut* __restrict__ C, int K) {
  const int nwc = COLS / 16;
  int lane = threadIdx.x & 31, wid = threadIdx.x >> 5;
  int wr = wid / nwc, wc = wid % nwc;
  int hf = lane >> 4, n = lane & 15;
  int row0 = blockIdx.x * 32 + wr * 16;
  int col0 = wc * 16;

  const __bf16* arow = A  + (size_t)(row0 + n) * K;   // A row M = lane&15
  const __bf16* brow = Wt + (size_t)(col0 + n) * K;   // B col N = lane&15

  v8f acc;
#pragma unroll
  for (int i = 0; i < 8; ++i) acc[i] = 0.f;

  for (int kb = 0; kb < K; kb += 32) {
    __builtin_prefetch(arow + kb + 128, 0, 0);        // global_prefetch_b8
    v16bf a = frag_a_bf16(arow, kb, hf);
    v16bf b = frag_b_bf16(brow + kb + hf * 16);
    acc = wmma_bf16(a, b, acc);
  }

#pragma unroll
  for (int r = 0; r < 8; ++r) {
    int row = row0 + r + hf * 8;                      // C layout: M = r (+8)
    int col = col0 + n;
    float x = acc[r] + bias[col];
    if (ACT == 1)      x = x / (1.f + __expf(-x));    // silu
    else if (ACT == 2) x = 1.f / (1.f + __expf(-x));  // sigmoid
    if constexpr (sizeof(TOut) == 2) C[(size_t)row * COLS + col] = f2bf(x);
    else                             C[(size_t)row * COLS + col] = x;
  }
}

// ---------------------------------------------------------------------------
// Per-token postprocess (bf16 in/out, fp32 math):
//   l2-normalize q,k rows in place; vt[b][d][l] = v*alpha + beta (transposed!)
// One wave per token, 2 elements per lane.
// ---------------------------------------------------------------------------
__global__ void __launch_bounds__(32)
rowpost(__bf16* __restrict__ q, __bf16* __restrict__ k,
        const __bf16* __restrict__ vin, __bf16* __restrict__ vt,
        const __bf16* __restrict__ alpha, const __bf16* __restrict__ beta) {
  size_t row = blockIdx.x;
  int lane = threadIdx.x;
  __bf16* qr = q + row * DDIM;
  __bf16* kr = k + row * DDIM;
  const __bf16* vr = vin   + row * DDIM;
  const __bf16* ar = alpha + row * DDIM;
  const __bf16* br = beta  + row * DDIM;

  float q0 = bf2f(qr[lane]), q1 = bf2f(qr[lane + 32]);
  float ss = wred_sum(q0 * q0 + q1 * q1);
  float qi = 1.f / fmaxf(sqrtf(ss), 1e-12f);
  qr[lane] = f2bf(q0 * qi); qr[lane + 32] = f2bf(q1 * qi);

  float k0 = bf2f(kr[lane]), k1 = bf2f(kr[lane + 32]);
  float ks = wred_sum(k0 * k0 + k1 * k1);
  float ki = 1.f / fmaxf(sqrtf(ks), 1e-12f);
  kr[lane] = f2bf(k0 * ki); kr[lane + 32] = f2bf(k1 * ki);

  // gated V, stored transposed: vt[(b*64 + d) * L + l]
  size_t b = row >> 11, l = row & (LSEQ - 1);
  __bf16* vtb = vt + (b * DDIM) * (size_t)LSEQ + l;
  vtb[(size_t)lane * LSEQ] =
      f2bf(bf2f(vr[lane]) * bf2f(ar[lane]) + bf2f(br[lane]));
  vtb[(size_t)(lane + 32) * LSEQ] =
      f2bf(bf2f(vr[lane + 32]) * bf2f(ar[lane + 32]) + bf2f(br[lane + 32]));
}

// ---------------------------------------------------------------------------
// Flash-attention style kernel. Grid: (L/32, B); block: 256 threads = 8 waves
// in a 2x4 layout (wr: query half, wc: key/head-dim quarter).
// Streams 64-key chunks, online softmax, bf16 WMMA for QK^T and P*V.
// V^T tile staged global->LDS via async-copy (ASYNCcnt) b128 instructions;
// all WMMA fragments are contiguous b128 loads.
// Epilogue: o/l -> rmsnorm * g_rms * shortcut_gate -> og (bf16).
// ---------------------------------------------------------------------------
__global__ void __launch_bounds__(256)
attn_kernel(const __bf16* __restrict__ q, const __bf16* __restrict__ k,
            const __bf16* __restrict__ vt, const __bf16* __restrict__ sc,
            const float* __restrict__ g_rms, __bf16* __restrict__ og) {
  __shared__ float  s_s[32][64];                      // scores / attn staging
  __shared__ __align__(16) __bf16 s_p[32][64];        // probabilities
  __shared__ __align__(16) __bf16 s_v[64][VPITCH];    // V^T chunk [dim][key]
  __shared__ float  s_m[32], s_l[32], s_cf[32];

  int tid = threadIdx.x, lane = tid & 31, wid = tid >> 5;
  int wr = wid >> 2, wc = wid & 3;
  int hf = lane >> 4, n = lane & 15;
  size_t base = (size_t)blockIdx.y * LSEQ;
  int q0 = blockIdx.x * 32;

  // LDS byte offset of s_v: low 32 bits of the flat shared address
  // (ISA 10.2, LDS aperture).
  unsigned ldsv = (unsigned)(size_t)(void*)&s_v[0][0];
  const __bf16* vtb = vt + (size_t)blockIdx.y * DDIM * LSEQ;

  // Q fragments held in registers for the whole kernel (D=64 -> 2 fragments).
  const __bf16* qrow = q + (base + q0 + wr * 16 + n) * DDIM;
  v16bf aq0 = frag_a_bf16(qrow, 0, hf);
  v16bf aq1 = frag_a_bf16(qrow, 32, hf);

  if (tid < 32) { s_m[tid] = -__builtin_inff(); s_l[tid] = 0.f; }

  v8f acc;
#pragma unroll
  for (int i = 0; i < 8; ++i) acc[i] = 0.f;

  for (int it = 0; it < LSEQ / 64; ++it) {
    __syncthreads();                              // prev P*V done; reuse LDS
    int kt = it * 64;

    // Stage V^T chunk [64 dims][64 keys] bf16 via async global->LDS b128:
    // 64 rows x 128B = 512 x 16B chunks, 2 per thread. Tracked by ASYNCcnt.
#pragma unroll
    for (int c = 0; c < 2; ++c) {
      int chunk = tid + c * 256;                  // 0..511
      int dd = chunk >> 3;                        // head-dim row
      int cc = chunk & 7;                         // 16B chunk in row
      const __bf16* src = vtb + (size_t)dd * LSEQ + kt + cc * 8;
      unsigned dst = ldsv + (unsigned)(dd * (VPITCH * 2) + cc * 16);
      asm volatile("global_load_async_to_lds_b128 %0, %1, off"
                   :: "v"(dst), "v"((unsigned long long)(const void*)src)
                   : "memory");
    }

    // QK^T: each wave computes a 16x16 score tile (2 WMMAs over D=64).
    v8f s;
#pragma unroll
    for (int i = 0; i < 8; ++i) s[i] = 0.f;
    const __bf16* krow = k + (base + kt + wc * 16 + n) * DDIM;
    s = wmma_bf16(aq0, frag_b_bf16(krow + hf * 16), s);
    s = wmma_bf16(aq1, frag_b_bf16(krow + 32 + hf * 16), s);
#pragma unroll
    for (int r = 0; r < 8; ++r) s_s[wr * 16 + r + hf * 8][wc * 16 + n] = s[r];

    asm volatile("s_wait_asynccnt 0" ::: "memory"); // own V copies done
    __syncthreads();                                // all waves: V + scores

    // Wave-parallel online softmax: wave w owns rows 4w..4w+3,
    // each lane covers 2 columns, ds_swizzle reductions.
#pragma unroll
    for (int rr = 0; rr < 4; ++rr) {
      int row = wid * 4 + rr;
      float x0 = s_s[row][lane] * 0.125f;
      float x1 = s_s[row][lane + 32] * 0.125f;
      float mx = wred_max(fmaxf(x0, x1));
      float mo = s_m[row];
      float mnew = fmaxf(mo, mx);
      float p0 = __expf(x0 - mnew), p1 = __expf(x1 - mnew);
      s_p[row][lane] = f2bf(p0);
      s_p[row][lane + 32] = f2bf(p1);
      float sum = wred_sum(p0 + p1);
      if (lane == 0) {
        float cf = __expf(mo - mnew);               // exp(-inf) = 0 on iter 0
        s_cf[row] = cf;
        s_l[row] = s_l[row] * cf + sum;
        s_m[row] = mnew;
      }
    }
    __syncthreads();

    // Rescale accumulator, then P*V (2 WMMAs per wave over 64 keys).
#pragma unroll
    for (int r = 0; r < 8; ++r) acc[r] *= s_cf[wr * 16 + r + hf * 8];
    int prow = wr * 16 + n;
    int hd = wc * 16 + n;
#pragma unroll
    for (int kc = 0; kc < 64; kc += 32) {
      v16bf pa = frag_a_bf16(&s_p[prow][0], kc, hf);
      v16bf vb = frag_b_bf16(&s_v[hd][kc + hf * 16]);   // contiguous keys
      acc = wmma_bf16(pa, vb, acc);
    }
  }

  __syncthreads();
#pragma unroll
  for (int r = 0; r < 8; ++r) {
    int row = wr * 16 + r + hf * 8;
    s_s[row][wc * 16 + n] = acc[r] / s_l[row];
  }
  __syncthreads();

  // Wave-parallel RMSNorm * g_rms * shortcut gate (bf16 out, coalesced).
#pragma unroll
  for (int rr = 0; rr < 4; ++rr) {
    int row = wid * 4 + rr;
    float x0 = s_s[row][lane], x1 = s_s[row][lane + 32];
    float ms = wred_sum(x0 * x0 + x1 * x1);
    float rs = rsqrtf(ms * (1.f / 64.f) + 1e-6f);
    size_t tok = base + q0 + row;
    og[tok * DDIM + lane] =
        f2bf(x0 * rs * g_rms[lane] * bf2f(sc[tok * DDIM + lane]));
    og[tok * DDIM + lane + 32] =
        f2bf(x1 * rs * g_rms[lane + 32] * bf2f(sc[tok * DDIM + lane + 32]));
  }
}

// ---------------------------------------------------------------------------
// Host orchestration.
// ---------------------------------------------------------------------------
extern "C" void kernel_launch(void* const* d_in, const int* in_sizes, int n_in,
                              void* d_out, int out_size, void* d_ws, size_t ws_size,
                              hipStream_t stream) {
  (void)in_sizes; (void)n_in; (void)out_size; (void)ws_size;

  const float* hq  = (const float*)d_in[0];
  const float* hk  = (const float*)d_in[1];
  const float* hv  = (const float*)d_in[2];
  const float* hs  = (const float*)d_in[3];
  const float* Wq  = (const float*)d_in[4];  const float* bq  = (const float*)d_in[5];
  const float* Wk  = (const float*)d_in[6];  const float* bk  = (const float*)d_in[7];
  const float* Wv  = (const float*)d_in[8];  const float* bv  = (const float*)d_in[9];
  const float* Wa1 = (const float*)d_in[10]; const float* ba1 = (const float*)d_in[11];
  const float* Wa2 = (const float*)d_in[12]; const float* ba2 = (const float*)d_in[13];
  const float* Wb  = (const float*)d_in[14]; const float* bb  = (const float*)d_in[15];
  const float* Ws1 = (const float*)d_in[16]; const float* bs1 = (const float*)d_in[17];
  const float* Ws2 = (const float*)d_in[18]; const float* bs2 = (const float*)d_in[19];
  const float* Wo  = (const float*)d_in[20]; const float* bo  = (const float*)d_in[21];
  const float* g_rms = (const float*)d_in[22];
  float* out = (float*)d_out;

  // Workspace layout (all bf16). ~17 MB.
  const size_t NE = (size_t)NTOK * EDIM;   // 4M   (hidden staging, reused)
  const size_t ND = (size_t)NTOK * DDIM;   // 512K
  __bf16* w   = (__bf16*)d_ws;
  __bf16* hb  = w;                 // staged hidden (bf16), reused per input
  __bf16* qb  = hb  + NE;
  __bf16* kbf = qb  + ND;
  __bf16* vbf = kbf + ND;          // silu(v) row-major
  __bf16* vtb = vbf + ND;          // gated V, transposed [B][D][L]
  __bf16* scb = vtb + ND;
  __bf16* alb = scb + ND;
  __bf16* beb = alb + ND;
  __bf16* agb = beb + ND;          // rmsnorm(attn)*gate -> final GEMM A
  __bf16* t1  = agb + ND;          // [NTOK, 32] MLP hidden (reused)
  __bf16* Wt_q  = t1 + (size_t)NTOK * 32;
  __bf16* Wt_k  = Wt_q  + 64 * 512;
  __bf16* Wt_v  = Wt_k  + 64 * 512;
  __bf16* Wt_b  = Wt_v  + 64 * 512;
  __bf16* Wt_a1 = Wt_b  + 64 * 512;
  __bf16* Wt_s1 = Wt_a1 + 32 * 512;
  __bf16* Wt_a2 = Wt_s1 + 32 * 512;
  __bf16* Wt_s2 = Wt_a2 + 64 * 32;
  __bf16* Wt_o  = Wt_s2 + 64 * 32;

  // 1) Weights -> transposed bf16.
  wtrans<<<64, 128, 0, stream>>>(Wq,  Wt_q,  512, 64);
  wtrans<<<64, 128, 0, stream>>>(Wk,  Wt_k,  512, 64);
  wtrans<<<64, 128, 0, stream>>>(Wv,  Wt_v,  512, 64);
  wtrans<<<64, 128, 0, stream>>>(Wb,  Wt_b,  512, 64);
  wtrans<<<32, 128, 0, stream>>>(Wa1, Wt_a1, 512, 32);
  wtrans<<<32, 128, 0, stream>>>(Ws1, Wt_s1, 512, 32);
  wtrans<<<64,  32, 0, stream>>>(Wa2, Wt_a2,  32, 64);
  wtrans<<<64,  32, 0, stream>>>(Ws2, Wt_s2,  32, 64);
  wtrans<<<64,  64, 0, stream>>>(Wo,  Wt_o,   64, 64);

  // 2) Projections (WMMA GEMMs, fused activations); stream order lets the
  //    single hidden-bf16 staging buffer be reused across inputs.
  dim3 g(NTOK / 32);
  const int N4 = (int)(NE / 4), HG = (int)(NE / 4 / 256);
  hconv<<<HG, 256, 0, stream>>>(hq, hb, N4);
  proj_gemm<1, 64, __bf16><<<g, 256, 0, stream>>>(hb, Wt_q,  bq,  qb,  512);
  hconv<<<HG, 256, 0, stream>>>(hk, hb, N4);
  proj_gemm<1, 64, __bf16><<<g, 256, 0, stream>>>(hb, Wt_k,  bk,  kbf, 512);
  hconv<<<HG, 256, 0, stream>>>(hv, hb, N4);
  proj_gemm<1, 64, __bf16><<<g, 256, 0, stream>>>(hb, Wt_v,  bv,  vbf, 512);
  proj_gemm<2, 64, __bf16><<<g, 256, 0, stream>>>(hb, Wt_b,  bb,  beb, 512);
  proj_gemm<0, 32, __bf16><<<g, 128, 0, stream>>>(hb, Wt_a1, ba1, t1,  512);
  proj_gemm<2, 64, __bf16><<<g, 256, 0, stream>>>(t1, Wt_a2, ba2, alb,  32);
  hconv<<<HG, 256, 0, stream>>>(hs, hb, N4);
  proj_gemm<0, 32, __bf16><<<g, 128, 0, stream>>>(hb, Wt_s1, bs1, t1,  512);
  proj_gemm<2, 64, __bf16><<<g, 256, 0, stream>>>(t1, Wt_s2, bs2, scb,  32);

  // 3) l2norm(q,k); gated V written transposed for contiguous B fragments.
  rowpost<<<NTOK, 32, 0, stream>>>(qb, kbf, vbf, vtb, alb, beb);

  // 4) Flash attention + rmsnorm + shortcut gate.
  attn_kernel<<<dim3(LSEQ / 32, BBATCH), 256, 0, stream>>>(qb, kbf, vtb, scb,
                                                           g_rms, agb);

  // 5) Output projection (fp32 out).
  proj_gemm<0, 64, float><<<g, 256, 0, stream>>>(agb, Wt_o, bo, out, 64);
}
